// MPNNet_v2_32409823216191
// MI455X (gfx1250) — matrix operand order repaired and checked by reading, not compiled, not run on hardware
//
#include <hip/hip_runtime.h>
#include <cfloat>
#include <cstdint>

typedef __attribute__((ext_vector_type(16))) _Float16     v16h;
typedef __attribute__((ext_vector_type(8)))  float        v8f;
typedef __attribute__((ext_vector_type(4)))  unsigned int v4u;
typedef __attribute__((ext_vector_type(4)))  _Float16     v4h;

namespace {
constexpr int NN     = 8192;    // nodes
constexpr int EE     = 16384;   // edges
constexpr int BG     = 256;     // graphs
constexpr int NF     = 14;
constexpr int DIM    = 128;
constexpr int D3     = 384;     // 3*D (GRU gates)
constexpr int D4     = 512;     // 4*D (LSTM gates)
constexpr int D8     = 1024;    // 8*D
constexpr int DD     = DIM * DIM;   // 16384
constexpr int NSTEM  = 2048;
constexpr int NJB    = 1024;
constexpr int NJB2   = 2048;
constexpr int LEVELS = 12;
constexpr int S2S    = 3;
constexpr int NS2O   = 105;     // ns2 output width (padded to 128 rows in ws)
}

__device__ __forceinline__ float leakyf(float v) { return v > 0.0f ? v : 0.01f * v; }
__device__ __forceinline__ float sigmoidf(float v) { return 1.0f / (1.0f + __expf(-v)); }

// monotone float<->uint encoding for atomic segment-max
__device__ __forceinline__ unsigned int enc_f32(float f) {
    unsigned int u = __float_as_uint(f);
    return (u & 0x80000000u) ? ~u : (u | 0x80000000u);
}
__device__ __forceinline__ float dec_f32(unsigned int u) {
    unsigned int v = (u & 0x80000000u) ? (u & 0x7FFFFFFFu) : ~u;
    return __uint_as_float(v);
}

// A fragment (16x32 f16, CDNA5 layout): lane half kb in {0,8};
// lane's 16 halves = K offsets {kb+0..kb+7 paired} and {kb+16..kb+23} -> 2x b128
__device__ __forceinline__ v16h load_fragA(const _Float16* __restrict__ p, int kb) {
    v4u x0 = *(const v4u*)(p + kb);
    v4u x1 = *(const v4u*)(p + kb + 16);
    return __builtin_bit_cast(v16h, __builtin_shufflevector(x0, x1, 0, 1, 2, 3, 4, 5, 6, 7));
}
// B fragment (32x16 f16): lane half kb in {0,8}; K offsets kb..kb+7 and kb+16..kb+23 -> 2x b128
__device__ __forceinline__ v16h load_fragB(const _Float16* __restrict__ p, int kb) {
    v4u x0 = *(const v4u*)(p + kb);
    v4u x1 = *(const v4u*)(p + kb + 16);
    return __builtin_bit_cast(v16h, __builtin_shufflevector(x0, x1, 0, 1, 2, 3, 4, 5, 6, 7));
}

// ---------------------------------------------------------------------------
// NT GEMM: C[M,N] = act( A[MxK] * B[NxK]^T + bias[N] )
// Requirements: M % 32 == 0, K % 32 == 0, B buffer padded to ceil(N/32)*32 rows.
// One wave32 per 32x32 macro-tile (2x2 WMMA tiles), f16 in, f32 accum.
// ---------------------------------------------------------------------------
template <int ACT>
__global__ __launch_bounds__(256) void gemm_nt_wmma(
    const _Float16* __restrict__ A, const _Float16* __restrict__ Bm,
    const float* __restrict__ bias, float* __restrict__ C32,
    _Float16* __restrict__ C16, int M, int N, int K) {
    const int wave  = blockIdx.x * (blockDim.x >> 5) + (threadIdx.x >> 5);
    const int lane  = threadIdx.x & 31;
    const int tN2 = (N + 31) >> 5;
    const int tM2 = (M + 31) >> 5;
    if (wave >= tM2 * tN2) return;
    const int tm = (wave / tN2) << 1;   // in units of 16-row tiles
    const int tn = (wave % tN2) << 1;

    const int l15  = lane & 15;
    const int kbA = (lane < 16) ? 0 : 8;
    const int kbB = (lane < 16) ? 0 : 8;

    const _Float16* a0p = A + (size_t)(tm * 16 + l15) * (size_t)K;
    const _Float16* a1p = a0p + (size_t)16 * (size_t)K;
    const _Float16* b0p = Bm + (size_t)(tn * 16 + l15) * (size_t)K;
    const _Float16* b1p = b0p + (size_t)16 * (size_t)K;

    v8f acc00 = {}, acc01 = {}, acc10 = {}, acc11 = {};
    for (int k0 = 0; k0 < K; k0 += 32) {
        v16h a0 = load_fragA(a0p + k0, kbA);
        v16h a1 = load_fragA(a1p + k0, kbA);
        v16h b0 = load_fragB(b0p + k0, kbB);
        v16h b1 = load_fragB(b1p + k0, kbB);
        acc00 = __builtin_amdgcn_wmma_f32_16x16x32_f16(false, a0, false, b0, (short)0, acc00, false, false);
        acc01 = __builtin_amdgcn_wmma_f32_16x16x32_f16(false, a0, false, b1, (short)0, acc01, false, false);
        acc10 = __builtin_amdgcn_wmma_f32_16x16x32_f16(false, a1, false, b0, (short)0, acc10, false, false);
        acc11 = __builtin_amdgcn_wmma_f32_16x16x32_f16(false, a1, false, b1, (short)0, acc11, false, false);
    }

    // store 4 tiles; C layout per ISA: lane<16 -> M=v, lane>=16 -> M=8+v; N = lane&15
    const int mbase = tm * 16 + ((lane < 16) ? 0 : 8);
#pragma unroll
    for (int t = 0; t < 4; ++t) {
        const v8f& acc = (t == 0) ? acc00 : (t == 1) ? acc01 : (t == 2) ? acc10 : acc11;
        const int col = (tn + ((t & 1))) * 16 + l15;
        const int rb  = mbase + ((t >> 1) ? 16 : 0);
        if (col >= N) continue;
        const float bv = bias ? bias[col] : 0.0f;
#pragma unroll
        for (int v = 0; v < 8; ++v) {
            const int row = rb + v;
            if (row >= M) continue;
            float val = acc[v] + bv;
            if (ACT == 1) val = leakyf(val);
            const size_t idx = (size_t)row * (size_t)N + (size_t)col;
            if (C32) C32[idx] = val;
            if (C16) C16[idx] = (_Float16)val;
        }
    }
}

// ---------------------------------------------------------------------------
// Elementwise / gather / scatter kernels
// ---------------------------------------------------------------------------
__global__ void k_zero_f32(float* __restrict__ p, int n) {
    int i = blockIdx.x * blockDim.x + threadIdx.x;
    if (i < n) p[i] = 0.0f;
}

__global__ void k_f32_to_f16(const float* __restrict__ s, _Float16* __restrict__ d, int n) {
    int i = blockIdx.x * blockDim.x + threadIdx.x;
    if (i < n) d[i] = (_Float16)s[i];
}

// roott16[d*D + i] = root[i*D + d]  (so out @ root becomes NT gemm)
__global__ void k_transpose_root(const float* __restrict__ root, _Float16* __restrict__ rt) {
    int i = blockIdx.x * blockDim.x + threadIdx.x;
    if (i >= DD) return;
    int r = i / DIM, c = i % DIM;
    rt[c * DIM + r] = (_Float16)root[i];
}

// out = leaky(x @ lin0_w.T + lin0_b) ; also f16 copy
__global__ void k_lin0(const float* __restrict__ x, const float* __restrict__ w,
                       const float* __restrict__ b, float* __restrict__ s32,
                       _Float16* __restrict__ s16) {
    int i = blockIdx.x * blockDim.x + threadIdx.x;
    if (i >= NN * DIM) return;
    int n = i / DIM, d = i % DIM;
    float acc = b[d];
#pragma unroll
    for (int k = 0; k < NF; ++k) acc += x[n * NF + k] * w[d * NF + k];
    acc = leakyf(acc);
    s32[i] = acc;
    s16[i] = (_Float16)acc;
}

// h1 = leaky(edge_attr @ e1_w.T + e1_b)  -> f16
__global__ void k_edge_mlp1(const float* __restrict__ ea, const float* __restrict__ w,
                            const float* __restrict__ b, _Float16* __restrict__ h1) {
    int i = blockIdx.x * blockDim.x + threadIdx.x;
    if (i >= EE * 128) return;
    int e = i / 128, k = i % 128;
    float acc = b[k];
#pragma unroll
    for (int j = 0; j < 4; ++j) acc += ea[e * 4 + j] * w[k * 4 + j];
    h1[i] = (_Float16)leakyf(acc);
}

__global__ void k_edge_count(const int* __restrict__ dst, float* __restrict__ cnt) {
    int e = blockIdx.x * blockDim.x + threadIdx.x;
    if (e < EE) atomicAdd(&cnt[dst[e]], 1.0f);
}

// one wave per edge: msg = out[src] @ We_e ; atomic scatter-add to agg[dst]
// lane owns 4 consecutive outputs -> contiguous 64-bit f16 loads per lane
__global__ __launch_bounds__(256) void k_edge_msg(
    const _Float16* __restrict__ s16, const _Float16* __restrict__ We,
    const int* __restrict__ src, const int* __restrict__ dst, float* __restrict__ agg) {
    int e = blockIdx.x * (blockDim.x >> 5) + (threadIdx.x >> 5);
    int lane = threadIdx.x & 31;
    if (e >= EE) return;
    const _Float16* xp = s16 + (size_t)src[e] * DIM;
    const _Float16* wp = We + (size_t)e * DD + 4 * lane;
    float a0 = 0.f, a1 = 0.f, a2 = 0.f, a3 = 0.f;
    for (int i = 0; i < DIM; ++i) {
        float xi = (float)xp[i];
        v4h w = *(const v4h*)(wp + i * DIM);
        a0 += xi * (float)w[0];
        a1 += xi * (float)w[1];
        a2 += xi * (float)w[2];
        a3 += xi * (float)w[3];
    }
    float* ap = agg + (size_t)dst[e] * DIM + 4 * lane;
    atomicAdd(ap + 0, a0); atomicAdd(ap + 1, a1);
    atomicAdd(ap + 2, a2); atomicAdd(ap + 3, a3);
}

// m = leaky(agg/cnt + rootp + conv_b) -> f16
__global__ void k_node_m(const float* __restrict__ agg, const float* __restrict__ cnt,
                         const float* __restrict__ rootp, const float* __restrict__ cb,
                         _Float16* __restrict__ m16) {
    int i = blockIdx.x * blockDim.x + threadIdx.x;
    if (i >= NN * DIM) return;
    int n = i / DIM, d = i % DIM;
    float c = cnt[n]; if (c < 1.0f) c = 1.0f;
    m16[i] = (_Float16)leakyf(agg[i] / c + rootp[i] + cb[d]);
}

// GRU combine (torch gate order r,z,n); updates state in place
__global__ void k_gru(const float* __restrict__ gi, const float* __restrict__ gh,
                      float* __restrict__ s32, _Float16* __restrict__ s16) {
    int i = blockIdx.x * blockDim.x + threadIdx.x;
    if (i >= NN * DIM) return;
    int n = i / DIM, d = i % DIM;
    const float* gin = gi + (size_t)n * D3;
    const float* ghn = gh + (size_t)n * D3;
    float r = sigmoidf(gin[d] + ghn[d]);
    float z = sigmoidf(gin[DIM + d] + ghn[DIM + d]);
    float nn = tanhf(gin[2 * DIM + d] + r * ghn[2 * DIM + d]);
    float h = s32[i];
    float hn = (1.0f - z) * nn + z * h;
    s32[i] = hn;
    s16[i] = (_Float16)hn;
}

// gather rows of width 1024 (per_atom) by index
__global__ void k_gather1024(const _Float16* __restrict__ pa, const int* __restrict__ idx,
                             _Float16* __restrict__ dst, int rows) {
    int i = blockIdx.x * blockDim.x + threadIdx.x;
    if (i >= rows * D8) return;
    int r = i / D8, c = i % D8;
    dst[i] = pa[(size_t)idx[r] * D8 + c];
}

// jbond final: per pair, two dot products with nj2_w then mean
__global__ void k_jbond(const _Float16* __restrict__ t2, const float* __restrict__ w,
                        const float* __restrict__ b, float* __restrict__ out) {
    int p = blockIdx.x * blockDim.x + threadIdx.x;
    if (p >= NJB) return;
    float s = 0.0f;
#pragma unroll
    for (int r = 0; r < 2; ++r) {
        const _Float16* row = t2 + (size_t)(2 * p + r) * DIM;
        float acc = b[0];
        for (int k = 0; k < DIM; ++k) acc += (float)row[k] * w[k];
        s += acc;
    }
    out[p] = 0.5f * s;
}

// ------------------------- Set2Set kernels ---------------------------------
__global__ void k_lstm_gates(const float* __restrict__ q, const float* __restrict__ hh,
                             const float* __restrict__ wih, const float* __restrict__ whh,
                             const float* __restrict__ bih, const float* __restrict__ bhh,
                             float* __restrict__ g) {
    int i = blockIdx.x * blockDim.x + threadIdx.x;
    if (i >= BG * D4) return;
    int b = i / D4, j = i % D4;
    float acc = bih[j] + bhh[j];
    const float* qb = q + (size_t)b * 256;
    const float* wj = wih + (size_t)j * 256;
    for (int k = 0; k < 256; ++k) acc += qb[k] * wj[k];
    const float* hb = hh + (size_t)b * DIM;
    const float* vj = whh + (size_t)j * DIM;
    for (int k = 0; k < DIM; ++k) acc += hb[k] * vj[k];
    g[i] = acc;
}

__global__ void k_lstm_update(const float* __restrict__ g, float* __restrict__ cc,
                              float* __restrict__ hh) {
    int i = blockIdx.x * blockDim.x + threadIdx.x;
    if (i >= BG * DIM) return;
    int b = i / DIM, d = i % DIM;
    const float* gb = g + (size_t)b * D4;
    float i_ = sigmoidf(gb[d]);
    float f_ = sigmoidf(gb[DIM + d]);
    float g_ = tanhf(gb[2 * DIM + d]);
    float o_ = sigmoidf(gb[3 * DIM + d]);
    float c = f_ * cc[i] + i_ * g_;
    cc[i] = c;
    hh[i] = o_ * tanhf(c);
}

__global__ void k_seg_init(unsigned int* __restrict__ emax, float* __restrict__ esum) {
    int b = blockIdx.x * blockDim.x + threadIdx.x;
    if (b >= BG) return;
    emax[b] = enc_f32(-FLT_MAX);
    esum[b] = 0.0f;
}

__global__ void k_attn_e(const float* __restrict__ s32, const float* __restrict__ hh,
                         const int* __restrict__ batch, float* __restrict__ evec,
                         unsigned int* __restrict__ emax) {
    int n = blockIdx.x * blockDim.x + threadIdx.x;
    if (n >= NN) return;
    int b = batch[n];
    const float* sp = s32 + (size_t)n * DIM;
    const float* hp = hh + (size_t)b * DIM;
    float acc = 0.0f;
    for (int d = 0; d < DIM; ++d) acc += sp[d] * hp[d];
    evec[n] = acc;
    atomicMax(&emax[b], enc_f32(acc));
}

__global__ void k_attn_exp(const float* __restrict__ evec, const int* __restrict__ batch,
                           const unsigned int* __restrict__ emax, float* __restrict__ ex,
                           float* __restrict__ esum) {
    int n = blockIdx.x * blockDim.x + threadIdx.x;
    if (n >= NN) return;
    int b = batch[n];
    float e = expf(evec[n] - dec_f32(emax[b]));
    ex[n] = e;
    atomicAdd(&esum[b], e);
}

__global__ void k_attn_rvec(const float* __restrict__ s32, const float* __restrict__ ex,
                            const float* __restrict__ esum, const int* __restrict__ batch,
                            float* __restrict__ rvec) {
    int i = blockIdx.x * blockDim.x + threadIdx.x;
    if (i >= NN * DIM) return;
    int n = i / DIM, d = i % DIM;
    int b = batch[n];
    float a = ex[n] / esum[b];
    atomicAdd(&rvec[(size_t)b * DIM + d], a * s32[i]);
}

__global__ void k_qstar(const float* __restrict__ hh, const float* __restrict__ rvec,
                        float* __restrict__ q) {
    int i = blockIdx.x * blockDim.x + threadIdx.x;
    if (i >= BG * 256) return;
    int b = i / 256, k = i % 256;
    q[i] = (k < DIM) ? hh[(size_t)b * DIM + k] : rvec[(size_t)b * DIM + (k - DIM)];
}

__global__ void k_sout(const float* __restrict__ q, const float* __restrict__ w,
                       const float* __restrict__ b, float* __restrict__ out) {
    int i = blockIdx.x * blockDim.x + threadIdx.x;
    if (i >= BG * 2) return;
    int g = i / 2, j = i % 2;
    float acc = b[j];
    const float* qb = q + (size_t)g * 256;
    const float* wj = w + (size_t)j * 256;
    for (int k = 0; k < 256; ++k) acc += qb[k] * wj[k];
    out[i] = acc;
}

// ---------------------------------------------------------------------------
extern "C" void kernel_launch(void* const* d_in, const int* in_sizes, int n_in,
                              void* d_out, int out_size, void* d_ws, size_t ws_size,
                              hipStream_t stream) {
    (void)in_sizes; (void)n_in; (void)out_size; (void)ws_size;

    // ---- inputs (setup_inputs dict order) ----
    const float* x        = (const float*)d_in[0];
    const int*   eidx     = (const int*)d_in[1];
    const float* eattr    = (const float*)d_in[2];
    const int*   batch    = (const int*)d_in[3];
    const int*   stem_idx = (const int*)d_in[4];
    const int*   jb_idx   = (const int*)d_in[5];
    const float* lin0_w = (const float*)d_in[6],  *lin0_b = (const float*)d_in[7];
    const float* e1_w   = (const float*)d_in[8],  *e1_b   = (const float*)d_in[9];
    const float* e2_w   = (const float*)d_in[10], *e2_b   = (const float*)d_in[11];
    const float* root   = (const float*)d_in[12], *conv_b = (const float*)d_in[13];
    const float* gwih   = (const float*)d_in[14], *gwhh   = (const float*)d_in[15];
    const float* gbih   = (const float*)d_in[16], *gbhh   = (const float*)d_in[17];
    const float* lin1_w = (const float*)d_in[18], *lin1_b = (const float*)d_in[19];
    const float* lwih   = (const float*)d_in[20], *lwhh   = (const float*)d_in[21];
    const float* lbih   = (const float*)d_in[22], *lbhh   = (const float*)d_in[23];
    const float* lin3_w = (const float*)d_in[24], *lin3_b = (const float*)d_in[25];
    const float* ns1_w  = (const float*)d_in[26], *ns1_b  = (const float*)d_in[27];
    const float* ns2_w  = (const float*)d_in[28], *ns2_b  = (const float*)d_in[29];
    const float* nj1_w  = (const float*)d_in[30], *nj1_b  = (const float*)d_in[31];
    const float* nj2_w  = (const float*)d_in[32], *nj2_b  = (const float*)d_in[33];

    const int* src = eidx;
    const int* dst = eidx + EE;

    float* outF    = (float*)d_out;
    float* sout_p  = outF;                         // (B,2)
    float* stem_p  = outF + BG * 2;                // (NSTEM,105)
    float* jb_p    = stem_p + NSTEM * NS2O;        // (NJB,)

    // ---- workspace bump allocator ----
    char* wsp = (char*)d_ws;
    size_t off = 0;
    auto alloc = [&](size_t bytes) -> void* {
        void* p = wsp + off;
        off = (off + bytes + 255) & ~(size_t)255;
        return p;
    };
    _Float16* We16   = (_Float16*)alloc((size_t)EE * DD * 2);   // 512 MB (dominant)
    _Float16* e2w16  = (_Float16*)alloc((size_t)DD * 128 * 2);
    _Float16* h1_16  = (_Float16*)alloc((size_t)EE * 128 * 2);
    float*    s32    = (float*)alloc((size_t)NN * DIM * 4);     // out == h
    _Float16* s16    = (_Float16*)alloc((size_t)NN * DIM * 2);
    float*    agg    = (float*)alloc((size_t)NN * DIM * 4);
    float*    rootp  = (float*)alloc((size_t)NN * DIM * 4);
    _Float16* m16    = (_Float16*)alloc((size_t)NN * DIM * 2);
    float*    gi     = (float*)alloc((size_t)NN * D3 * 4);
    float*    gh     = (float*)alloc((size_t)NN * D3 * 4);
    float*    cnt    = (float*)alloc((size_t)NN * 4);
    _Float16* rt16   = (_Float16*)alloc((size_t)DD * 2);
    _Float16* gwih16 = (_Float16*)alloc((size_t)D3 * DIM * 2);
    _Float16* gwhh16 = (_Float16*)alloc((size_t)D3 * DIM * 2);
    _Float16* l1w16  = (_Float16*)alloc((size_t)D8 * DIM * 2);
    _Float16* pa16   = (_Float16*)alloc((size_t)NN * D8 * 2);
    _Float16* ns1w16 = (_Float16*)alloc((size_t)DIM * D8 * 2);
    _Float16* nj1w16 = (_Float16*)alloc((size_t)DIM * D8 * 2);
    _Float16* ns2w16 = (_Float16*)alloc((size_t)128 * DIM * 2);  // padded 105 -> 128 rows
    _Float16* stin16 = (_Float16*)alloc((size_t)NSTEM * D8 * 2);
    _Float16* t1_16  = (_Float16*)alloc((size_t)NSTEM * DIM * 2);
    _Float16* jbin16 = (_Float16*)alloc((size_t)NJB2 * D8 * 2);
    _Float16* t2_16  = (_Float16*)alloc((size_t)NJB2 * DIM * 2);
    float*    qstar  = (float*)alloc((size_t)BG * 256 * 4);
    float*    hh     = (float*)alloc((size_t)BG * DIM * 4);
    float*    cc     = (float*)alloc((size_t)BG * DIM * 4);
    float*    gbuf   = (float*)alloc((size_t)BG * D4 * 4);
    float*    evec   = (float*)alloc((size_t)NN * 4);
    float*    exv    = (float*)alloc((size_t)NN * 4);
    unsigned int* emax = (unsigned int*)alloc((size_t)BG * 4);
    float*    esum   = (float*)alloc((size_t)BG * 4);
    float*    rvec   = (float*)alloc((size_t)BG * DIM * 4);

    auto g1 = [](long long n, int bs) { return dim3((unsigned)((n + bs - 1) / bs)); };
    auto zero = [&](float* p, long long n) {
        k_zero_f32<<<g1(n, 256), 256, 0, stream>>>(p, (int)n);
    };
    auto cvt = [&](const float* s, _Float16* d, long long n) {
        k_f32_to_f16<<<g1(n, 256), 256, 0, stream>>>(s, d, (int)n);
    };
    auto gemm = [&](const _Float16* A, const _Float16* Bm, const float* bias,
                    float* C32, _Float16* C16, int M, int N, int K, int act) {
        int tiles = ((M + 31) / 32) * ((N + 31) / 32);
        int blocks = (tiles + 7) / 8;  // 8 waves (of 32) per 256-thread block
        if (act)
            gemm_nt_wmma<1><<<blocks, 256, 0, stream>>>(A, Bm, bias, C32, C16, M, N, K);
        else
            gemm_nt_wmma<0><<<blocks, 256, 0, stream>>>(A, Bm, bias, C32, C16, M, N, K);
    };

    // ---- weight prep (f32 -> f16, + root transpose; ns2 padded w/ zeros) ----
    cvt(e2_w, e2w16, (long long)DD * 128);
    cvt(gwih, gwih16, (long long)D3 * DIM);
    cvt(gwhh, gwhh16, (long long)D3 * DIM);
    cvt(lin1_w, l1w16, (long long)D8 * DIM);
    cvt(ns1_w, ns1w16, (long long)DIM * D8);
    cvt(nj1_w, nj1w16, (long long)DIM * D8);
    zero((float*)ns2w16, (long long)128 * DIM / 2);   // zero 128x128 f16 pad region
    cvt(ns2_w, ns2w16, (long long)NS2O * DIM);
    k_transpose_root<<<g1(DD, 256), 256, 0, stream>>>(root, rt16);

    // ---- lin0 + edge MLP stage 1 ----
    k_lin0<<<g1((long long)NN * DIM, 256), 256, 0, stream>>>(x, lin0_w, lin0_b, s32, s16);
    k_edge_mlp1<<<g1((long long)EE * 128, 256), 256, 0, stream>>>(eattr, e1_w, e1_b, h1_16);

    // ---- We = h1 @ e2_w.T + e2_b  (M=16384, N=16384, K=128; WMMA, f16 out) ----
    gemm(h1_16, e2w16, e2_b, nullptr, We16, EE, DD, 128, /*act=*/0);

    // ---- in-degree counts ----
    zero(cnt, NN);
    k_edge_count<<<g1(EE, 256), 256, 0, stream>>>(dst, cnt);

    // ---- 12 message-passing levels ----
    for (int lvl = 0; lvl < LEVELS; ++lvl) {
        zero(agg, (long long)NN * DIM);
        k_edge_msg<<<g1((long long)EE * 32, 256), 256, 0, stream>>>(s16, We16, src, dst, agg);
        gemm(s16, rt16, nullptr, rootp, nullptr, NN, DIM, DIM, 0);          // out @ root
        k_node_m<<<g1((long long)NN * DIM, 256), 256, 0, stream>>>(agg, cnt, rootp, conv_b, m16);
        gemm(m16, gwih16, gbih, gi, nullptr, NN, D3, DIM, 0);               // gi
        gemm(s16, gwhh16, gbhh, gh, nullptr, NN, D3, DIM, 0);               // gh
        k_gru<<<g1((long long)NN * DIM, 256), 256, 0, stream>>>(gi, gh, s32, s16);
    }

    // ---- per_atom = leaky(out @ lin1_w.T + lin1_b) (f16) ----
    gemm(s16, l1w16, lin1_b, nullptr, pa16, NN, D8, DIM, /*act=*/1);

    // ---- stem head ----
    k_gather1024<<<g1((long long)NSTEM * D8, 256), 256, 0, stream>>>(pa16, stem_idx, stin16, NSTEM);
    gemm(stin16, ns1w16, ns1_b, nullptr, t1_16, NSTEM, DIM, D8, /*act=*/1);
    gemm(t1_16, ns2w16, ns2_b, stem_p, nullptr, NSTEM, NS2O, DIM, /*act=*/0);

    // ---- jbond head ----
    k_gather1024<<<g1((long long)NJB2 * D8, 256), 256, 0, stream>>>(pa16, jb_idx, jbin16, NJB2);
    gemm(jbin16, nj1w16, nj1_b, nullptr, t2_16, NJB2, DIM, D8, /*act=*/1);
    k_jbond<<<g1(NJB, 256), 256, 0, stream>>>(t2_16, nj2_w, nj2_b, jb_p);

    // ---- Set2Set (3 steps) ----
    zero(qstar, (long long)BG * 256);
    zero(hh, (long long)BG * DIM);
    zero(cc, (long long)BG * DIM);
    for (int s = 0; s < S2S; ++s) {
        k_lstm_gates<<<g1((long long)BG * D4, 256), 256, 0, stream>>>(
            qstar, hh, lwih, lwhh, lbih, lbhh, gbuf);
        k_lstm_update<<<g1((long long)BG * DIM, 256), 256, 0, stream>>>(gbuf, cc, hh);
        k_seg_init<<<g1(BG, 256), 256, 0, stream>>>(emax, esum);
        zero(rvec, (long long)BG * DIM);
        k_attn_e<<<g1(NN, 256), 256, 0, stream>>>(s32, hh, batch, evec, emax);
        k_attn_exp<<<g1(NN, 256), 256, 0, stream>>>(evec, batch, emax, exv, esum);
        k_attn_rvec<<<g1((long long)NN * DIM, 256), 256, 0, stream>>>(s32, exv, esum, batch, rvec);
        k_qstar<<<g1((long long)BG * 256, 256), 256, 0, stream>>>(hh, rvec, qstar);
    }
    k_sout<<<g1(BG * 2, 256), 256, 0, stream>>>(qstar, lin3_w, lin3_b, sout_p);
}
